// Mult_CA_60885456388399
// MI455X (gfx1250) — matrix operand order repaired and checked by reading, not compiled, not run on hardware
//
#include <hip/hip_runtime.h>

// ---------------------------------------------------------------------------
// MI455X (gfx1250) fused cross-attention.
// All GEMMs via v_wmma_f32_16x16x32_bf16 (wave32, fp32 accumulate).
// The N x C attention map (536MB fp32) is NEVER materialized: S is computed
// twice (online column-softmax stats pass + fused attn*V pass), keeping HBM
// traffic at ~100MB total vs. >1GB for the naive path (23.3 TB/s -> ~5-10us).
// Round 2: 16x64 per-wave tiles in the projection GEMMs (4 independent WMMA
// chains per A-fragment -> hides WMMA->VALU hazards, 4x A-operand reuse) and
// global_prefetch on the two streaming loops.
// ---------------------------------------------------------------------------

typedef __attribute__((ext_vector_type(16))) __bf16 bf16x16;
typedef __attribute__((ext_vector_type(8)))  __bf16 bf16x8;
typedef __attribute__((ext_vector_type(8)))  float  f32x8;

#define WMMA_BF16(A, B, C)                                                     \
  __builtin_amdgcn_wmma_f32_16x16x32_bf16(false, (A), false, (B), (short)0,    \
                                          (C), false, false)

static constexpr int DIM    = 256;
static constexpr int DIMC   = 1024;   // DIM*COEF
static constexpr int H      = 32;
static constexpr int DH     = 32;
static constexpr int C_TOK  = 256;
static constexpr int N_TOK  = 16384;
static constexpr float EPS  = 1e-9f;

// ---- fragment loaders (per CDNA5 ISA 7.12.2 wave32 layouts) ----------------
// A (16x32, 16-bit): lane l -> row (l&15); K runs [8*(l>>4), +8) and +16.
__device__ __forceinline__ bf16x16 load_a_bf16(const __bf16* p) {
  bf16x8 lo = *(const bf16x8*)(p);
  bf16x8 hi = *(const bf16x8*)(p + 16);
  bf16x16 a;
#pragma unroll
  for (int e = 0; e < 8; ++e) { a[e] = lo[e]; a[8 + e] = hi[e]; }
  return a;
}

__device__ __forceinline__ bf16x16 load_a_f32(const float* p) {
  f32x8 lo = *(const f32x8*)(p);
  f32x8 hi = *(const f32x8*)(p + 16);
  bf16x16 a;
#pragma unroll
  for (int e = 0; e < 8; ++e) {
    a[e]     = (__bf16)lo[e];
    a[8 + e] = (__bf16)hi[e];
  }
  return a;
}

// B (32x16, 16-bit) from a TRANSPOSED (col-major) operand Bt[n][k]:
// lane l -> column n0+(l&15); K runs [k0+8*(l>>4), +8) and +16 (same as A).
__device__ __forceinline__ bf16x16 load_b(const __bf16* bt, int ld, int n0,
                                          int k0) {
  int l = threadIdx.x & 31;
  const __bf16* p = bt + (size_t)(n0 + (l & 15)) * ld + k0 + ((l >> 4) << 3);
  return load_a_bf16(p);
}

__device__ __forceinline__ const __bf16* a_ptr(const __bf16* a, int ld, int m0,
                                               int k0) {
  int l = threadIdx.x & 31;
  return a + (size_t)(m0 + (l & 15)) * ld + k0 + ((l >> 4) << 3);
}

// D (16x16 f32): lane l, VGPR r -> row r+8*(l>>4), col l&15.

// ---------------------------------------------------------------------------
// Kernel 1: bf16 transpose of weights: out[c][r] = (bf16) w[r][c]
__global__ __launch_bounds__(256) void wtrans_kernel(const float* __restrict__ w,
                                                     __bf16* __restrict__ out,
                                                     int R, int Cc) {
  int idx = blockIdx.x * blockDim.x + threadIdx.x;
  if (idx >= R * Cc) return;
  int c = idx / R, r = idx - c * R;
  out[idx] = (__bf16)w[(size_t)r * Cc + c];
}

// ---------------------------------------------------------------------------
// Kernel 2: K/V projection. Each wave: 16 rows x 64 cols (4 WMMA chains per
// A-fragment). transposed=0 -> out[h][c][d], transposed=1 -> out[h][d][c].
__global__ __launch_bounds__(256) void proj_kv_kernel(
    const float* __restrict__ sem, const __bf16* __restrict__ WT,
    const float* __restrict__ bias, __bf16* __restrict__ out, int transposed) {
  int wave = threadIdx.x >> 5, l = threadIdx.x & 31;
  int flat = blockIdx.x * 8 + wave;        // 256 wave-jobs
  int c0 = (flat >> 4) * 16;               // 16 c-tiles
  int jg = (flat & 15) * 64;               // 16 j-groups of 64
  f32x8 acc[4] = {};
#pragma unroll
  for (int kk = 0; kk < 8; ++kk) {
    int k0 = kk * 32;
    bf16x16 a = load_a_f32(sem + (size_t)(c0 + (l & 15)) * DIM + k0 +
                           ((l >> 4) << 3));
#pragma unroll
    for (int jt = 0; jt < 4; ++jt) {
      bf16x16 b = load_b(WT, DIM, jg + jt * 16, k0);
      acc[jt] = WMMA_BF16(a, b, acc[jt]);
    }
  }
#pragma unroll
  for (int jt = 0; jt < 4; ++jt) {
    int j0 = jg + jt * 16;
    float bb = bias[j0 + (l & 15)];
    int h = j0 >> 5, d = (j0 & 31) + (l & 15);
#pragma unroll
    for (int r = 0; r < 8; ++r) {
      int c = c0 + r + ((l >> 4) << 3);
      float v = acc[jt][r] + bb;
      if (!transposed)
        out[((size_t)h * C_TOK + c) * DH + d] = (__bf16)v;
      else
        out[((size_t)h * DH + d) * C_TOK + c] = (__bf16)v;
    }
  }
}

// ---------------------------------------------------------------------------
// Kernel 3: Q projection -> Qh[h][n][d] bf16. 16x64 per wave.
__global__ __launch_bounds__(256) void proj_q_kernel(
    const float* __restrict__ geo, const __bf16* __restrict__ WqT,
    const float* __restrict__ bq, __bf16* __restrict__ Qh) {
  int wave = threadIdx.x >> 5, l = threadIdx.x & 31;
  int flat = blockIdx.x * 8 + wave;        // 16384 wave-jobs
  int n0 = (flat >> 4) * 16;               // 1024 n-tiles
  int jg = (flat & 15) * 64;               // 16 j-groups of 64
  f32x8 acc[4] = {};
#pragma unroll
  for (int kk = 0; kk < 8; ++kk) {
    int k0 = kk * 32;
    bf16x16 a = load_a_f32(geo + (size_t)(n0 + (l & 15)) * DIM + k0 +
                           ((l >> 4) << 3));
#pragma unroll
    for (int jt = 0; jt < 4; ++jt) {
      bf16x16 b = load_b(WqT, DIM, jg + jt * 16, k0);
      acc[jt] = WMMA_BF16(a, b, acc[jt]);
    }
  }
#pragma unroll
  for (int jt = 0; jt < 4; ++jt) {
    int j0 = jg + jt * 16;
    float bb = bq[j0 + (l & 15)];
    int h = j0 >> 5, d = (j0 & 31) + (l & 15);
#pragma unroll
    for (int r = 0; r < 8; ++r) {
      int n = n0 + r + ((l >> 4) << 3);
      Qh[((size_t)h * N_TOK + n) * DH + d] = (__bf16)(acc[jt][r] + bb);
    }
  }
}

// ---------------------------------------------------------------------------
// Kernel 4: online column-softmax stats. Each wave owns 16 columns of one
// head, keeps K's B-fragment in registers, streams Q A-fragments over a 2048
// row chunk (with prefetch 4 tiles ahead). Partial (m, Z) per chunk.
__global__ __launch_bounds__(512) void col_stats_kernel(
    const __bf16* __restrict__ Qh, const __bf16* __restrict__ Kh,
    float* __restrict__ pstats) {
  int h = blockIdx.y, chunk = blockIdx.x;   // 32 x 8
  int wave = threadIdx.x >> 5, l = threadIdx.x & 31;
  int c0 = wave * 16;
  const __bf16* kbase = Kh + (size_t)h * C_TOK * DH;
  const __bf16* qbase = Qh + (size_t)h * N_TOK * DH;
  bf16x16 bk = load_b(kbase, DH, c0, 0);    // resident across the whole loop
  float m = -3.0e38f, Z = 0.f;
  for (int t = 0; t < 128; ++t) {
    int n0 = chunk * 2048 + t * 16;
    const __bf16* ap = a_ptr(qbase, DH, n0, 0);
    __builtin_prefetch(ap + 64 * DH, 0, 0);   // 4 n-tiles ahead
    bf16x16 a = load_a_bf16(ap);
    f32x8 z = {};
    f32x8 s = WMMA_BF16(a, bk, z);
    float tmax = s[0];
#pragma unroll
    for (int r = 1; r < 8; ++r) tmax = fmaxf(tmax, s[r]);
    float mn = fmaxf(m, tmax);
    float esum = 0.f;
#pragma unroll
    for (int r = 0; r < 8; ++r) esum += __expf(s[r] - mn);
    Z = Z * __expf(m - mn) + esum;
    m = mn;
  }
  // merge lane l with lane l^16 (same column, other row-half)
  float mo = __shfl_xor(m, 16, 32);
  float Zo = __shfl_xor(Z, 16, 32);
  float M = fmaxf(m, mo);
  float Zc = Z * __expf(m - M) + Zo * __expf(mo - M);
  if (l < 16) {
    float* p = pstats + (((size_t)h * 8 + chunk) * C_TOK + (c0 + l)) * 2;
    p[0] = M;
    p[1] = Zc;
  }
}

// Kernel 5: merge the 8 chunk partials; store m and 1/Z.
__global__ __launch_bounds__(256) void stats_combine_kernel(
    const float* __restrict__ pstats, float* __restrict__ stats) {
  int idx = blockIdx.x * blockDim.x + threadIdx.x;
  if (idx >= H * C_TOK) return;
  int h = idx >> 8, c = idx & 255;
  float m = -3.0e38f, Z = 0.f;
  for (int ch = 0; ch < 8; ++ch) {
    const float* p = pstats + (((size_t)h * 8 + ch) * C_TOK + c) * 2;
    float M = fmaxf(m, p[0]);
    Z = Z * __expf(m - M) + p[1] * __expf(p[0] - M);
    m = M;
  }
  float* q = stats + ((size_t)h * C_TOK + c) * 2;
  q[0] = m;
  q[1] = (Z > 0.f) ? 1.f / Z : 0.f;
}

// ---------------------------------------------------------------------------
// Kernel 6: fused S-recompute + column-softmax + L1 row-norm + attn*V.
// Wave handles 16 rows of one head. Unnormalized e staged in LDS (bf16) only
// to re-fragment D-layout -> A-layout; the 1/(EPS+rowsum) scale is applied
// directly on the x D-accumulators (rows align with the lane's D rows).
__global__ __launch_bounds__(128) void attn_v_kernel(
    const __bf16* __restrict__ Qh, const __bf16* __restrict__ Kh,
    const __bf16* __restrict__ Vt, const float* __restrict__ stats,
    __bf16* __restrict__ Xbuf) {
  __shared__ __align__(16) __bf16 elds[4][16][C_TOK];   // 32 KB
  int h = blockIdx.y;
  int wave = threadIdx.x >> 5, l = threadIdx.x & 31;
  int n0 = (blockIdx.x * 4 + wave) * 16;
  const __bf16* qbase = Qh + (size_t)h * N_TOK * DH;
  const __bf16* kbase = Kh + (size_t)h * C_TOK * DH;
  const __bf16* vbase = Vt + (size_t)h * DH * C_TOK;
  const float* st = stats + (size_t)h * C_TOK * 2;

  bf16x16 aq = load_a_bf16(a_ptr(qbase, DH, n0, 0));    // Q fragment, reused
  float rsum[8];
#pragma unroll
  for (int r = 0; r < 8; ++r) rsum[r] = 0.f;

#pragma unroll
  for (int ct = 0; ct < 16; ++ct) {
    int c0 = ct * 16;
    bf16x16 bkf = load_b(kbase, DH, c0, 0);
    f32x8 z = {};
    f32x8 s = WMMA_BF16(aq, bkf, z);
    int c = c0 + (l & 15);
    float mc = st[c * 2], rZ = st[c * 2 + 1];
#pragma unroll
    for (int r = 0; r < 8; ++r) {
      float e = __expf(s[r] - mc) * rZ;
      rsum[r] += e;
      elds[wave][r + ((l >> 4) << 3)][c] = (__bf16)e;
    }
  }
  __syncthreads();

  // full row sums: butterfly over the 16 lanes sharing the same row half
#pragma unroll
  for (int off = 1; off < 16; off <<= 1)
#pragma unroll
    for (int r = 0; r < 8; ++r) rsum[r] += __shfl_xor(rsum[r], off, 32);

  f32x8 x0 = {}, x1 = {};
#pragma unroll
  for (int cc = 0; cc < 8; ++cc) {
    int k0 = cc * 32;
    bf16x16 ae = load_a_bf16(&elds[wave][l & 15][k0 + ((l >> 4) << 3)]);
    bf16x16 b0 = load_b(vbase, C_TOK, 0, k0);
    bf16x16 b1 = load_b(vbase, C_TOK, 16, k0);
    x0 = WMMA_BF16(ae, b0, x0);
    x1 = WMMA_BF16(ae, b1, x1);
  }
#pragma unroll
  for (int r = 0; r < 8; ++r) {
    float inv = 1.f / (EPS + rsum[r]);
    int n = n0 + r + ((l >> 4) << 3);
    __bf16* xp = Xbuf + (size_t)n * DIMC + h * DH;
    xp[l & 15]        = (__bf16)(x0[r] * inv);
    xp[16 + (l & 15)] = (__bf16)(x1[r] * inv);
  }
}

// ---------------------------------------------------------------------------
// Kernel 7: output projection, fp32 result + bias. 16x64 per wave (4 chains),
// prefetch on the streaming X operand.
__global__ __launch_bounds__(256) void out_proj_kernel(
    const __bf16* __restrict__ Xbuf, const __bf16* __restrict__ WpT,
    const float* __restrict__ bp, float* __restrict__ out) {
  int wave = threadIdx.x >> 5, l = threadIdx.x & 31;
  int flat = blockIdx.x * 8 + wave;        // 4096 wave-jobs
  int n0 = (flat >> 2) * 16;               // 1024 n-tiles
  int jg = (flat & 3) * 64;                // 4 j-groups of 64
  f32x8 acc[4] = {};
  for (int kk = 0; kk < 32; ++kk) {
    int k0 = kk * 32;
    const __bf16* ap = a_ptr(Xbuf, DIMC, n0, k0);
    __builtin_prefetch(ap + 128, 0, 0);    // 4 K-steps ahead in the row
    bf16x16 a = load_a_bf16(ap);
#pragma unroll
    for (int jt = 0; jt < 4; ++jt) {
      bf16x16 b = load_b(WpT, DIMC, jg + jt * 16, k0);
      acc[jt] = WMMA_BF16(a, b, acc[jt]);
    }
  }
#pragma unroll
  for (int jt = 0; jt < 4; ++jt) {
    int j0 = jg + jt * 16;
    float bb = bp[j0 + (l & 15)];
#pragma unroll
    for (int r = 0; r < 8; ++r)
      out[(size_t)(n0 + r + ((l >> 4) << 3)) * DIM + j0 + (l & 15)] =
          acc[jt][r] + bb;
  }
}

// ---------------------------------------------------------------------------
extern "C" void kernel_launch(void* const* d_in, const int* in_sizes, int n_in,
                              void* d_out, int out_size, void* d_ws,
                              size_t ws_size, hipStream_t stream) {
  const float* sem = (const float*)d_in[0];
  const float* geo = (const float*)d_in[1];
  const float* Wk  = (const float*)d_in[2];
  const float* bk  = (const float*)d_in[3];
  const float* Wq  = (const float*)d_in[4];
  const float* bq  = (const float*)d_in[5];
  const float* Wv  = (const float*)d_in[6];
  const float* bv  = (const float*)d_in[7];
  const float* Wp  = (const float*)d_in[8];
  const float* bp  = (const float*)d_in[9];
  float* out = (float*)d_out;

  char* ws = (char*)d_ws;
  constexpr size_t OFF_WQT   = 0;                       // 1024x256 bf16
  constexpr size_t OFF_WKT   = OFF_WQT + 524288;
  constexpr size_t OFF_WVT   = OFF_WKT + 524288;
  constexpr size_t OFF_WPT   = OFF_WVT + 524288;        // 256x1024 bf16
  constexpr size_t OFF_KH    = OFF_WPT + 524288;        // [32][256][32] bf16
  constexpr size_t OFF_VT    = OFF_KH + 524288;         // [32][32][256] bf16
  constexpr size_t OFF_QH    = OFF_VT + 524288;         // [32][16384][32] bf16
  constexpr size_t OFF_XBUF  = OFF_QH + 33554432;       // [16384][1024] bf16
  constexpr size_t OFF_PSTAT = OFF_XBUF + 33554432;     // [32][8][256][2] f32
  constexpr size_t OFF_STAT  = OFF_PSTAT + 524288;      // [32][256][2] f32

  __bf16* WqT  = (__bf16*)(ws + OFF_WQT);
  __bf16* WkT  = (__bf16*)(ws + OFF_WKT);
  __bf16* WvT  = (__bf16*)(ws + OFF_WVT);
  __bf16* WpT  = (__bf16*)(ws + OFF_WPT);
  __bf16* KhB  = (__bf16*)(ws + OFF_KH);
  __bf16* VtB  = (__bf16*)(ws + OFF_VT);
  __bf16* QhB  = (__bf16*)(ws + OFF_QH);
  __bf16* Xbuf = (__bf16*)(ws + OFF_XBUF);
  float* pstat = (float*)(ws + OFF_PSTAT);
  float* stat  = (float*)(ws + OFF_STAT);

  wtrans_kernel<<<1024, 256, 0, stream>>>(Wq, WqT, DIM, DIMC);
  wtrans_kernel<<<1024, 256, 0, stream>>>(Wk, WkT, DIM, DIMC);
  wtrans_kernel<<<1024, 256, 0, stream>>>(Wv, WvT, DIM, DIMC);
  wtrans_kernel<<<1024, 256, 0, stream>>>(Wp, WpT, DIMC, DIM);

  proj_kv_kernel<<<32, 256, 0, stream>>>(sem, WkT, bk, KhB, 0);
  proj_kv_kernel<<<32, 256, 0, stream>>>(sem, WvT, bv, VtB, 1);
  proj_q_kernel<<<2048, 256, 0, stream>>>(geo, WqT, bq, QhB);

  col_stats_kernel<<<dim3(8, 32), 512, 0, stream>>>(QhB, KhB, pstat);
  stats_combine_kernel<<<32, 256, 0, stream>>>(pstat, stat);

  attn_v_kernel<<<dim3(256, 32), 128, 0, stream>>>(QhB, KhB, VtB, stat, Xbuf);
  out_proj_kernel<<<512, 256, 0, stream>>>(Xbuf, WpT, bp, out);
}
